// RMamba2TimeSeriesDirect_30064771072278
// MI455X (gfx1250) — matrix operand order repaired and checked by reading, not compiled, not run on hardware
//
#include <hip/hip_runtime.h>
#include <hip/hip_bf16.h>
#include <math.h>

// Mamba2 time-series forward for gfx1250, specialized for the reference's
// dataflow: only hidden[:, -1, :] feeds the output, so Y_diag/Y_off run on the
// last chunk only; the inter-chunk state recurrence covers all chunks.
// All GEMMs use v_wmma_f32_16x16x32_bf16 (fp32 accumulate). LDS tiles are laid
// out so every WMMA fragment is 2x ds_load_b128 (A: [m][k] rows, two 8-elem
// K-runs at klo/klo+16; B: [n][k] rows, two 8-elem K-runs at kb/kb+16) per ISA 7.12.2.

typedef __attribute__((ext_vector_type(16))) __bf16         v16bf;
typedef __attribute__((ext_vector_type(16))) unsigned short v16us;
typedef __attribute__((ext_vector_type(8)))  float          v8f;

__device__ __forceinline__ unsigned short f2bf(float f) {
  unsigned u = __float_as_uint(f);
  u += 0x7fffu + ((u >> 16) & 1u);            // round-to-nearest-even
  return (unsigned short)(u >> 16);
}
__device__ __forceinline__ float bf2f(unsigned short h) {
  return __uint_as_float(((unsigned)h) << 16);
}
__device__ __forceinline__ float siluf(float x) { return x / (1.f + __expf(-x)); }
__device__ __forceinline__ float softplusf(float x) { return x > 20.f ? x : log1pf(__expf(x)); }

// D = A(16x32 bf16) * B(32x16 bf16) + C(16x16 f32)
__device__ __forceinline__ v8f wmma_bf16(v16us a, v16us b, v8f c) {
  return __builtin_amdgcn_wmma_f32_16x16x32_bf16(
      false, __builtin_bit_cast(v16bf, a),
      false, __builtin_bit_cast(v16bf, b),
      (short)0, c, false, false);
}

// A fragment from an [m][k]-major LDS row (row byte address 16B-aligned):
// lanes 0-15 hold K {0..7, 16..23}; lanes 16-31 hold K {8..15, 24..31}.
__device__ __forceinline__ v16us frag_a(const unsigned short* row, int lane) {
  int klo = (lane >> 4) << 3;
  union { uint4 u[2]; v16us v; } r;
  r.u[0] = *(const uint4*)(row + klo);
  r.u[1] = *(const uint4*)(row + klo + 16);
  return r.v;
}
// B fragment from an [n][k]-major LDS row: lanes 0-15 K {0..7, 16..23}; lanes 16-31 K {8..15, 24..31}.
__device__ __forceinline__ v16us frag_b(const unsigned short* row, int lane) {
  int kb = (lane >> 4) << 3;
  union { uint4 u[2]; v16us v; } r;
  r.u[0] = *(const uint4*)(row + kb);
  r.u[1] = *(const uint4*)(row + kb + 16);
  return r.v;
}

// ---------------------------------------------------------------- kernel 1
// u[b,l,d] = (pad? 0 : inputs[b,seg,1+t,d]) + pe[seg,d]; fp32 + bf16 copies
__global__ void k_prep(const float* __restrict__ inp, float* __restrict__ u32,
                       unsigned short* __restrict__ u16) {
  long i = (long)blockIdx.x * 256 + threadIdx.x;   // < 16384*192
  int d = (int)(i % 192);
  long row = i / 192;
  int l = (int)(row % 2048), b = (int)(row / 2048);
  int seg = l >> 8, t = l & 255;
  float v = 0.f;
  if (t < 128) v = inp[(((long)(b * 8 + seg) * 129) + 1 + t) * 192 + d];
  int ii = d >> 1;
  float div = __expf((float)(2 * ii) * (-9.210340371976184f / 192.f));
  float ang = (float)seg * div;
  v += (d & 1) ? __cosf(ang) : __sinf(ang);
  u32[i] = v;
  u16[i] = f2bf(v);
}

// ---------------------------------------------------------------- kernel 2
// bf16 TRANSPOSED cast of in_proj_w columns [1536,3352): wT[n][k], n<1856 padded
__global__ void k_wcast(const float* __restrict__ w, unsigned short* __restrict__ wbT) {
  long i = (long)blockIdx.x * 256 + threadIdx.x;   // < 1856*192
  int k = (int)(i % 192), n = (int)(i / 192);
  float v = (n < 1816) ? w[(long)k * 3352 + 1536 + n] : 0.f;
  wbT[i] = f2bf(v);
}

// ---------------------------------------------------------------- kernel 3
// GEMM: [16384 x 192] x [192 x 1856] -> xBC(1792 cols) + dt-pre(24 cols)
// block tile 64x64, 8 waves, each wave: one 16x16 M-tile x two 16-col N-tiles
__global__ __launch_bounds__(256) void k_gemm_inproj(
    const unsigned short* __restrict__ A, const unsigned short* __restrict__ BT,
    float* __restrict__ Cout) {
  extern __shared__ char smem[];
  unsigned short* As = (unsigned short*)smem;   // [m=64][k=32] stride 40
  unsigned short* Bt = As + 64 * 40;            // [n=64][k=32] stride 40
  const int tid = threadIdx.x, lane = tid & 31, w = tid >> 5;
  const int m0 = blockIdx.x * 64, n0 = blockIdx.y * 64;
  const int mw = (w >> 1) * 16, nw = (w & 1) * 32;
  v8f acc0 = {}, acc1 = {};
  const int r8 = tid >> 2, c8 = (tid & 3) * 8;   // 64 rows x 4 chunks of 8
  for (int k0 = 0; k0 < 192; k0 += 32) {
    *(uint4*)(As + r8 * 40 + c8) = *(const uint4*)(A  + (long)(m0 + r8) * 192 + k0 + c8);
    *(uint4*)(Bt + r8 * 40 + c8) = *(const uint4*)(BT + (long)(n0 + r8) * 192 + k0 + c8);
    if (k0 + 32 < 192) {
      __builtin_prefetch(A  + (long)(m0 + r8) * 192 + k0 + 32 + c8, 0, 1);
      __builtin_prefetch(BT + (long)(n0 + r8) * 192 + k0 + 32 + c8, 0, 1);
    }
    __syncthreads();
    v16us af  = frag_a(As + (mw + (lane & 15)) * 40, lane);
    v16us bf0 = frag_b(Bt + (nw      + (lane & 15)) * 40, lane);
    v16us bf1 = frag_b(Bt + (nw + 16 + (lane & 15)) * 40, lane);
    acc0 = wmma_bf16(af, bf0, acc0);
    acc1 = wmma_bf16(af, bf1, acc1);
    __syncthreads();
  }
  const int rofs = (lane >> 4) << 3, cof = lane & 15;
#pragma unroll
  for (int r = 0; r < 8; ++r) {
    long gm = m0 + mw + r + rofs;
    Cout[gm * 1856 + n0 + nw + cof]      = acc0[r];
    Cout[gm * 1856 + n0 + nw + 16 + cof] = acc1[r];
  }
}

// ---------------------------------------------------------------- kernel 4
// dt = softplus(g + dt_bias); per-(b,c,h) inclusive cumsum of dt*A over chunk
__global__ void k_dt_scan(const float* __restrict__ G, const float* __restrict__ dt_bias,
                          const float* __restrict__ A_log, float* __restrict__ dt,
                          float* __restrict__ Acs) {
  int lane = threadIdx.x & 31;
  int gid = blockIdx.x * 8 + (threadIdx.x >> 5);   // (b*8+c)*24+h
  int h = gid % 24, c = (gid / 24) % 8, b = gid / 192;
  float An = -__expf(A_log[h]);
  float bias = dt_bias[h];
  float carry = 0.f;
  long rowbase = (long)b * 2048 + c * 256;
  for (int it = 0; it < 8; ++it) {
    int l = it * 32 + lane;
    long row = rowbase + l;
    float dv = softplusf(G[row * 1856 + 1792 + h] + bias);
    dt[row * 24 + h] = dv;
    float v = dv * An;
#pragma unroll
    for (int off = 1; off < 32; off <<= 1) {
      float n = __shfl_up(v, off, 32);
      if (lane >= off) v += n;
    }
    Acs[(long)gid * 256 + l] = carry + v;
    carry += __shfl(v, 31, 32);
  }
}

// ---------------------------------------------------------------- kernel 5
// causal depthwise conv (k=4) + SiLU; emit X = xs*dt (bf16), B, C (bf16),
// and xs fp32 at the final position
__global__ void k_conv(const float* __restrict__ G, const float* __restrict__ cw,
                       const float* __restrict__ cb, const float* __restrict__ dt,
                       unsigned short* __restrict__ Xb, unsigned short* __restrict__ Bb,
                       unsigned short* __restrict__ Cb, float* __restrict__ xs_last) {
  long i = (long)blockIdx.x * 256 + threadIdx.x;   // < 8*2048*1792
  int ch = (int)(i % 1792);
  long row = i / 1792;
  int l = (int)(row % 2048), b = (int)(row / 2048);
  float acc = cb[ch];
#pragma unroll
  for (int k = 0; k < 4; ++k) {
    int lq = l + k - 3;
    if (lq >= 0) acc += G[((long)b * 2048 + lq) * 1856 + ch] * cw[ch * 4 + k];
  }
  float v = siluf(acc);
  long grow = (long)b * 2048 + l;
  if (ch < 1536) {
    int h = ch >> 6;
    Xb[grow * 1536 + ch] = f2bf(v * dt[grow * 24 + h]);
    if (l == 2047) xs_last[b * 1536 + ch] = v;
  } else if (ch < 1664) {
    Bb[grow * 128 + (ch - 1536)] = f2bf(v);
  } else {
    Cb[grow * 128 + (ch - 1664)] = f2bf(v);
  }
}

// ---------------------------------------------------------------- kernel 6
// last-chunk scores S = C * B^T : [256x128]x[128x256] per batch.
// Bb is stored [s][n] = exactly the [N][K] major order the B fragment wants.
__global__ __launch_bounds__(256) void k_scores7(const unsigned short* __restrict__ Cb,
                                                 const unsigned short* __restrict__ Bb,
                                                 float* __restrict__ S) {
  extern __shared__ char smem[];
  unsigned short* As = (unsigned short*)smem;   // [l=64][k=32] stride 40
  unsigned short* Bt = As + 64 * 40;            // [s=64][k=32] stride 40
  int tid = threadIdx.x, lane = tid & 31, w = tid >> 5;
  int m0 = blockIdx.x * 64, n0 = blockIdx.y * 64, b = blockIdx.z;
  long base = (long)b * 2048 + 1792;            // chunk 7
  int mw = (w >> 1) * 16, nw = (w & 1) * 32;
  v8f acc0 = {}, acc1 = {};
  int r8 = tid >> 2, c8 = (tid & 3) * 8;
  for (int k0 = 0; k0 < 128; k0 += 32) {
    *(uint4*)(As + r8 * 40 + c8) = *(const uint4*)(Cb + (base + m0 + r8) * 128 + k0 + c8);
    *(uint4*)(Bt + r8 * 40 + c8) = *(const uint4*)(Bb + (base + n0 + r8) * 128 + k0 + c8);
    __syncthreads();
    v16us af  = frag_a(As + (mw + (lane & 15)) * 40, lane);
    v16us bf0 = frag_b(Bt + (nw      + (lane & 15)) * 40, lane);
    v16us bf1 = frag_b(Bt + (nw + 16 + (lane & 15)) * 40, lane);
    acc0 = wmma_bf16(af, bf0, acc0);
    acc1 = wmma_bf16(af, bf1, acc1);
    __syncthreads();
  }
  int rofs = (lane >> 4) << 3, cof = lane & 15;
#pragma unroll
  for (int r = 0; r < 8; ++r) {
    long gm = (long)b * 256 + m0 + mw + r + rofs;
    S[gm * 256 + n0 + nw + cof]      = acc0[r];
    S[gm * 256 + n0 + nw + 16 + cof] = acc1[r];
  }
}

// ---------------------------------------------------------------- kernel 7
// chunk states: St[p,n] = sum_l X[l,p] * exp(Atot - Acs[l]) * B[l,n]
// one block per (b,c,h); output 64x128, K = 256 over l.
// LDS: Xs as [p][l] (A-major), Bs as [n][l] (B-major, decay folded in).
__global__ __launch_bounds__(256) void k_states(const unsigned short* __restrict__ Xb,
                                                const unsigned short* __restrict__ Bb,
                                                const float* __restrict__ Acs,
                                                float* __restrict__ St) {
  extern __shared__ char smem[];
  unsigned short* Xs = (unsigned short*)smem;   // [p=64][l=32] stride 40
  unsigned short* Bs = Xs + 64 * 40;            // [n=128][l=32] stride 40
  int tid = threadIdx.x, lane = tid & 31, w = tid >> 5;
  int idx = blockIdx.x;                         // (b*8+c)*24+h
  int h = idx % 24, c = (idx / 24) % 8, b = idx / 192;
  long rowbase = (long)b * 2048 + c * 256;
  float Atot = Acs[(long)idx * 256 + 255];
  int mw = (w & 3) * 16, nbase = (w >> 2) * 64;
  v8f acc[4] = {};
  int ll = tid >> 3, xp = (tid & 7) * 8, ng = (tid & 7) * 16;
  for (int l0 = 0; l0 < 256; l0 += 32) {
    {   // X tile: transpose [l][p] -> [p][l]
      __align__(16) unsigned short tmp[8];
      *(uint4*)tmp = *(const uint4*)(Xb + (rowbase + l0 + ll) * 1536 + h * 64 + xp);
#pragma unroll
      for (int q = 0; q < 8; ++q) Xs[(xp + q) * 40 + ll] = tmp[q];
    }
    {   // B tile: transpose [l][n] -> [n][l], fold per-row decay
      float dec = __expf(Atot - Acs[(long)idx * 256 + l0 + ll]);
      __align__(16) unsigned short tmp[16];
      *(uint4*)(tmp)     = *(const uint4*)(Bb + (rowbase + l0 + ll) * 128 + ng);
      *(uint4*)(tmp + 8) = *(const uint4*)(Bb + (rowbase + l0 + ll) * 128 + ng + 8);
#pragma unroll
      for (int q = 0; q < 16; ++q) Bs[(ng + q) * 40 + ll] = f2bf(bf2f(tmp[q]) * dec);
    }
    __syncthreads();
    v16us af = frag_a(Xs + (mw + (lane & 15)) * 40, lane);
#pragma unroll
    for (int j = 0; j < 4; ++j) {
      v16us bf = frag_b(Bs + (nbase + j * 16 + (lane & 15)) * 40, lane);
      acc[j] = wmma_bf16(af, bf, acc[j]);
    }
    __syncthreads();
  }
  int rofs = (lane >> 4) << 3, cof = lane & 15;
#pragma unroll
  for (int j = 0; j < 4; ++j)
#pragma unroll
    for (int r = 0; r < 8; ++r)
      St[((long)idx * 64 + mw + r + rofs) * 128 + nbase + j * 16 + cof] = acc[j][r];
}

// ---------------------------------------------------------------- kernel 8
// inter-chunk recurrence; emit state entering chunk 7 transposed [n][p] bf16
__global__ void k_scan(const float* __restrict__ St, const float* __restrict__ Acs,
                       unsigned short* __restrict__ st0) {
  int bh = blockIdx.x;                // b*24+h
  int b = bh / 24, h = bh % 24;
  int t = threadIdx.x;
  float carry[32];
#pragma unroll
  for (int j = 0; j < 32; ++j) carry[j] = 0.f;
  for (int c = 0; c < 7; ++c) {
    long idx = ((long)b * 8 + c) * 24 + h;
    float dec = __expf(Acs[idx * 256 + 255]);
    const float* s = St + idx * 8192 + (long)t * 32;
#pragma unroll
    for (int j = 0; j < 32; ++j) carry[j] = carry[j] * dec + s[j];
  }
#pragma unroll
  for (int j = 0; j < 32; ++j) {
    int e = t * 32 + j, p = e >> 7, n = e & 127;
    st0[(((long)b * 24 + h) * 128 + n) * 64 + p] = f2bf(carry[j]);
  }
}

// ---------------------------------------------------------------- kernel 9
// last-chunk Y = (tril(exp(Acs[l]-Acs[s])) o S) @ X + (exp(Acs[l]) o C) @ st0^T
// one block per (b,h); both GEMMs accumulate into the same v8f accumulators.
// LDS: Ts [l=256][k=32] (A-major), Xs [p=64][k=32] (B-major).
__global__ __launch_bounds__(256) void k_ylast(const float* __restrict__ S,
                                               const unsigned short* __restrict__ Xb,
                                               const unsigned short* __restrict__ Cb,
                                               const unsigned short* __restrict__ st0,
                                               const float* __restrict__ Acs,
                                               float* __restrict__ Y) {
  extern __shared__ char smem[];
  float* acss = (float*)smem;                          // 256 f32
  unsigned short* Ts = (unsigned short*)(acss + 256);  // [l=256][k=32] stride 40
  unsigned short* Xs = Ts + 256 * 40;                  // [p=64][k=32] stride 40
  int tid = threadIdx.x, lane = tid & 31, w = tid >> 5;
  int bh = blockIdx.x;
  int b = bh / 24, h = bh % 24;
  long aidx = (((long)b * 8 + 7) * 24 + h) * 256;
  acss[tid] = Acs[aidx + tid];
  __syncthreads();
  long base = (long)b * 2048 + 1792;
  v8f acc[2][4] = {};
  int ll = tid >> 3, pg = (tid & 7) * 8;
  // phase 1: Y_diag over s
  for (int s0 = 0; s0 < 256; s0 += 32) {
    float al = acss[tid];
#pragma unroll
    for (int ss = 0; ss < 32; ++ss) {
      int sg = s0 + ss;
      float v = (sg <= tid) ? __expf(al - acss[sg]) * S[((long)b * 256 + tid) * 256 + sg] : 0.f;
      Ts[tid * 40 + ss] = f2bf(v);
    }
    {   // X tile: transpose [s][p] -> [p][s]
      __align__(16) unsigned short tmp[8];
      *(uint4*)tmp = *(const uint4*)(Xb + (base + s0 + ll) * 1536 + h * 64 + pg);
#pragma unroll
      for (int q = 0; q < 8; ++q) Xs[(pg + q) * 40 + ll] = tmp[q];
    }
    __syncthreads();
#pragma unroll
    for (int mt = 0; mt < 2; ++mt) {
      v16us af = frag_a(Ts + (w * 32 + mt * 16 + (lane & 15)) * 40, lane);
#pragma unroll
      for (int j = 0; j < 4; ++j) {
        v16us bf = frag_b(Xs + (j * 16 + (lane & 15)) * 40, lane);
        acc[mt][j] = wmma_bf16(af, bf, acc[mt][j]);
      }
    }
    __syncthreads();
  }
  // phase 2: Y_off over state dim n
  long sbase = ((long)b * 24 + h) * 128;
  for (int n0 = 0; n0 < 128; n0 += 32) {
    float sc = __expf(acss[tid]);
#pragma unroll
    for (int nn = 0; nn < 32; ++nn)
      Ts[tid * 40 + nn] = f2bf(bf2f(Cb[(base + tid) * 128 + n0 + nn]) * sc);
    {   // st0 tile: transpose [n][p] -> [p][n]
      __align__(16) unsigned short tmp[8];
      *(uint4*)tmp = *(const uint4*)(st0 + (sbase + n0 + ll) * 64 + pg);
#pragma unroll
      for (int q = 0; q < 8; ++q) Xs[(pg + q) * 40 + ll] = tmp[q];
    }
    __syncthreads();
#pragma unroll
    for (int mt = 0; mt < 2; ++mt) {
      v16us af = frag_a(Ts + (w * 32 + mt * 16 + (lane & 15)) * 40, lane);
#pragma unroll
      for (int j = 0; j < 4; ++j) {
        v16us bf = frag_b(Xs + (j * 16 + (lane & 15)) * 40, lane);
        acc[mt][j] = wmma_bf16(af, bf, acc[mt][j]);
      }
    }
    __syncthreads();
  }
  int rofs = (lane >> 4) << 3, cof = lane & 15;
#pragma unroll
  for (int mt = 0; mt < 2; ++mt)
#pragma unroll
    for (int j = 0; j < 4; ++j)
#pragma unroll
      for (int r = 0; r < 8; ++r) {
        int l = w * 32 + mt * 16 + r + rofs;
        Y[(((long)b * 256 + l) * 24 + h) * 64 + j * 16 + cof] = acc[mt][j][r];
      }
}

// ---------------------------------------------------------------- kernel 10
// z at the last position only: z[b,d] = u[b,2047,:] . in_proj_w[:, d]
__global__ void k_zlast(const float* __restrict__ u32, const float* __restrict__ W,
                        float* __restrict__ zl) {
  int i = blockIdx.x * 256 + threadIdx.x;   // < 8*1536
  int d = i % 1536, b = i / 1536;
  const float* ur = u32 + ((long)b * 2048 + 2047) * 192;
  float a = 0.f;
  for (int k = 0; k < 192; ++k) a += ur[k] * W[(long)k * 3352 + d];
  zl[i] = a;
}

// ---------------------------------------------------------------- kernel 11
// gate, RMSNorm, classifier, broadcast store (one block per batch)
__global__ __launch_bounds__(256) void k_final(const float* __restrict__ Y,
                                               const float* __restrict__ xsl,
                                               const float* __restrict__ zl,
                                               const float* __restrict__ Dp,
                                               const float* __restrict__ nw,
                                               const float* __restrict__ clsw,
                                               const float* __restrict__ clsb,
                                               float* __restrict__ out) {
  __shared__ float yv[1536];
  __shared__ float red[8];
  __shared__ float lg[48];
  int b = blockIdx.x, t = threadIdx.x, lane = t & 31, w = t >> 5;
  float local = 0.f;
  for (int d = t; d < 1536; d += 256) {
    int h = d >> 6, p = d & 63;
    float y = Y[(((long)b * 256 + 255) * 24 + h) * 64 + p] + xsl[b * 1536 + d] * Dp[h];
    float z = zl[b * 1536 + d];
    y *= z / (1.f + __expf(-z));
    yv[d] = y;
    local += y * y;
  }
#pragma unroll
  for (int o = 16; o > 0; o >>= 1) local += __shfl_down(local, o, 32);
  if (lane == 0) red[w] = local;
  __syncthreads();
  if (t == 0) {
    float s2 = 0.f;
    for (int i2 = 0; i2 < 8; ++i2) s2 += red[i2];
    red[0] = rsqrtf(s2 / 1536.f + 1e-5f);
  }
  __syncthreads();
  float rms = red[0];
  for (int d = t; d < 1536; d += 256) yv[d] *= rms * nw[d];
  __syncthreads();
  if (t < 48) {
    float a = clsb[t];
    for (int d = 0; d < 1536; ++d) a += yv[d] * clsw[d * 48 + t];
    lg[t] = a;
  }
  __syncthreads();
  for (int i = t; i < 2048 * 48; i += 256) out[(long)b * 2048 * 48 + i] = lg[i % 48];
}

// ---------------------------------------------------------------- launch
extern "C" void kernel_launch(void* const* d_in, const int* in_sizes, int n_in,
                              void* d_out, int out_size, void* d_ws, size_t ws_size,
                              hipStream_t stream) {
  (void)in_sizes; (void)n_in; (void)out_size; (void)ws_size;
  const float* inp   = (const float*)d_in[0];
  const float* inw   = (const float*)d_in[1];
  const float* convw = (const float*)d_in[2];
  const float* convb = (const float*)d_in[3];
  const float* dtb   = (const float*)d_in[4];
  const float* alog  = (const float*)d_in[5];
  const float* Dp    = (const float*)d_in[6];
  const float* nw    = (const float*)d_in[7];
  // d_in[8] = out_proj_w: its product is discarded by the reference -> skipped
  const float* clsw  = (const float*)d_in[9];
  const float* clsb  = (const float*)d_in[10];
  float* out = (float*)d_out;

  char* ws = (char*)d_ws;
  size_t off = 0;
  auto take = [&](size_t bytes) -> char* {
    char* p = ws + off;
    off = (off + bytes + 255) & ~(size_t)255;
    return p;
  };
  float*          u32  = (float*)take((size_t)16384 * 192 * 4);
  unsigned short* u16  = (unsigned short*)take((size_t)16384 * 192 * 2);
  unsigned short* wbT  = (unsigned short*)take((size_t)1856 * 192 * 2);
  float*          gemm = (float*)take((size_t)16384 * 1856 * 4);
  float*          dt   = (float*)take((size_t)16384 * 24 * 4);
  float*          acs  = (float*)take((size_t)1536 * 256 * 4);
  unsigned short* Xb   = (unsigned short*)take((size_t)16384 * 1536 * 2);
  unsigned short* Bb   = (unsigned short*)take((size_t)16384 * 128 * 2);
  unsigned short* Cb   = (unsigned short*)take((size_t)16384 * 128 * 2);
  float*          xsl  = (float*)take((size_t)8 * 1536 * 4);
  float*          Smat = (float*)take((size_t)8 * 256 * 256 * 4);
  float*          St   = (float*)take((size_t)1536 * 8192 * 4);
  unsigned short* st0  = (unsigned short*)take((size_t)192 * 8192 * 2);
  float*          Y    = (float*)take((size_t)8 * 256 * 24 * 64 * 4);
  float*          zl   = (float*)take((size_t)8 * 1536 * 4);

  k_prep<<<12288, 256, 0, stream>>>(inp, u32, u16);
  k_wcast<<<1392, 256, 0, stream>>>(inw, wbT);
  k_gemm_inproj<<<dim3(256, 29), 256, (64 * 40 + 64 * 40) * 2, stream>>>(u16, wbT, gemm);
  k_dt_scan<<<192, 256, 0, stream>>>(gemm, dtb, alog, dt, acs);
  k_conv<<<114688, 256, 0, stream>>>(gemm, convw, convb, dt, Xb, Bb, Cb, xsl);
  k_scores7<<<dim3(4, 4, 8), 256, (64 * 40 + 64 * 40) * 2, stream>>>(Cb, Bb, Smat);
  k_states<<<1536, 256, (64 * 40 + 128 * 40) * 2, stream>>>(Xb, Bb, acs, St);
  k_scan<<<192, 256, 0, stream>>>(St, acs, st0);
  k_ylast<<<192, 256, 256 * 4 + (256 * 40 + 64 * 40) * 2, stream>>>(Smat, Xb, Cb, st0, acs, Y);
  k_zlast<<<48, 256, 0, stream>>>(u32, inw, zl);
  k_final<<<8, 256, 0, stream>>>(Y, xsl, zl, Dp, nw, clsw, clsb, out);
}
